// Adpt_Bias_Denoiser_65884798321327
// MI455X (gfx1250) — matrix operand order repaired and checked by reading, not compiled, not run on hardware
//
#include <hip/hip_runtime.h>
#include <stdint.h>
#include <math.h>

// Problem constants
#define B_      16
#define T_      600
#define S_      600
#define D_      1024
#define H_      8
#define L_      8
#define DFF_    4096
#define IN_     52
#define PERIOD_ 25
#define HD_     128
#define TA_     601   // T (or S) + 1 adapter row

typedef __attribute__((ext_vector_type(16))) __bf16 v16bf;
typedef __attribute__((ext_vector_type(2)))  __bf16 v2bf;
typedef __attribute__((ext_vector_type(8)))  float  v8f;

union U16 { uint32_t u[8]; v16bf v; };

// pack two f32 into a bf16 pair (lo in [15:0], hi in [31:16]) -- single
// v_cvt_pk_bf16_f32; bit_cast (not union) so the packed reg flows straight
// into the DS store without element extraction.
static __device__ __forceinline__ uint32_t pk2(float lo, float hi) {
#if __has_builtin(__builtin_amdgcn_cvt_pk_bf16_f32)
  return __builtin_bit_cast(uint32_t, __builtin_amdgcn_cvt_pk_bf16_f32(lo, hi));
#else
  v2bf t; t[0] = (__bf16)lo; t[1] = (__bf16)hi;
  return __builtin_bit_cast(uint32_t, t);
#endif
}

// ---------------------------------------------------------------------------
// WMMA GEMM: C[M,N] = act(A[M,K] @ W[N,K]^T + bias[N])
// 128x128 block tile, 8 waves (4 x 2), wave tile 32x64 (2x4 fragments), BK=32.
// Double-buffered LDS (bf16 pairs as u32), register prefetch, branchless
// clamped loads (garbage rows never stored).
// ---------------------------------------------------------------------------
#define BM 128
#define BN 128
#define BK 32
#define LPADU 18   // u32 per LDS row: 16 pairs + 2 pad

__global__ __launch_bounds__(256)
void gemm_bf16(const float* __restrict__ A, const float* __restrict__ W,
               const float* __restrict__ bias, float* __restrict__ C,
               int M, int N, int K, int act) {
  __shared__ uint32_t As[2][BM * LPADU];
  __shared__ uint32_t Bs[2][BN * LPADU];
  const int tid  = threadIdx.x;
  const int lane = tid & 31;
  const int wave = tid >> 5;
  const int wm = wave >> 1;       // 0..3 : M direction
  const int wn = wave & 1;        // 0..1 : N direction
  const int m0 = blockIdx.y * BM;
  const int n0 = blockIdx.x * BN;
  const int lo = lane & 15;
  const int hi = lane >> 4;

  v8f acc[2][4];
#pragma unroll
  for (int i = 0; i < 2; i++)
#pragma unroll
    for (int j = 0; j < 4; j++) acc[i][j] = (v8f)0.f;

  const int row2 = tid >> 1;          // 0..127
  const int col0 = (tid & 1) * 16;    // float col 0 or 16
  int am = m0 + row2; if (am > M - 1) am = M - 1;   // clamp: no divergence
  int bn = n0 + row2; if (bn > N - 1) bn = N - 1;
  const float* aP = A + (size_t)am * K + col0;
  const float* bP = W + (size_t)bn * K + col0;

  float4 ra[4], rb[4];
#pragma unroll
  for (int i = 0; i < 4; i++) {
    ra[i] = ((const float4*)aP)[i];
    rb[i] = ((const float4*)bP)[i];
  }

  int buf = 0;
  const int sb = row2 * LPADU + (col0 >> 1);
  for (int k0 = 0; k0 < K; k0 += BK) {
#pragma unroll
    for (int i = 0; i < 4; i++) {
      As[buf][sb + 2 * i]     = pk2(ra[i].x, ra[i].y);
      As[buf][sb + 2 * i + 1] = pk2(ra[i].z, ra[i].w);
      Bs[buf][sb + 2 * i]     = pk2(rb[i].x, rb[i].y);
      Bs[buf][sb + 2 * i + 1] = pk2(rb[i].z, rb[i].w);
    }
    __syncthreads();
    if (k0 + BK < K) {
      const float* ap2 = aP + k0 + BK;
      const float* bp2 = bP + k0 + BK;
#pragma unroll
      for (int i = 0; i < 4; i++) {
        ra[i] = ((const float4*)ap2)[i];
        rb[i] = ((const float4*)bp2)[i];
      }
    }
    v16bf a[2], b[4];
#pragma unroll
    for (int i = 0; i < 2; i++) {
      U16 t;
      int base = (wm * 32 + i * 16 + lo) * LPADU;
#pragma unroll
      for (int v = 0; v < 4; v++) {
        t.u[v]     = As[buf][base + hi * 4 + v];
        t.u[4 + v] = As[buf][base + 8 + hi * 4 + v];
      }
      a[i] = t.v;
    }
#pragma unroll
    for (int j = 0; j < 4; j++) {
      U16 t;
      int base = (wn * 64 + j * 16 + lo) * LPADU;
#pragma unroll
      for (int v = 0; v < 8; v++) t.u[v] = Bs[buf][base + (v >> 2) * 8 + hi * 4 + (v & 3)];
      b[j] = t.v;
    }
#pragma unroll
    for (int i = 0; i < 2; i++)
#pragma unroll
      for (int j = 0; j < 4; j++)
        acc[i][j] = __builtin_amdgcn_wmma_f32_16x16x32_bf16(
            false, a[i], false, b[j], (short)0, acc[i][j], false, false);
    buf ^= 1;
  }

#pragma unroll
  for (int j = 0; j < 4; j++) {
    int n = n0 + wn * 64 + j * 16 + lo;
    if (n >= N) continue;
    float bv = bias ? bias[n] : 0.f;
#pragma unroll
    for (int i = 0; i < 2; i++) {
#pragma unroll
      for (int r = 0; r < 8; r++) {
        int m = m0 + wm * 32 + i * 16 + r + hi * 8;
        if (m < M) {
          float val = acc[i][j][r] + bv;
          if (act == 1) val = fmaxf(val, 0.f);
          C[(size_t)m * N + n] = val;
        }
      }
    }
  }
}

// ---------------------------------------------------------------------------
// Self-attention (adapter + analytic alibi). One WG per (16-q tile, head, b).
// ---------------------------------------------------------------------------
#define NKT   38     // ceil(601/16)
#define SCOLS 608    // 601 padded to 19*32

__global__ __launch_bounds__(256)
void self_attn(const float* __restrict__ qkv, float* __restrict__ out) {
  // qs: Q tile (16 rows x 64 pairs + pad -> 68). vt: V chunk transposed,
  // double buffered: [2][128 dims x 17 pairs]. Union: phases are disjoint.
  __shared__ union {
    uint32_t qs[16 * 68];
    uint32_t vt[2][128 * 17];
  } uqv;
  // s: scores f32 [16][608]; p: bf16 pairs written in place (wave-private rows)
  __shared__ union {
    float    s[16 * SCOLS];
    uint32_t p[16 * SCOLS];   // row r pairs live at [r*SCOLS .. r*SCOLS+303]
  } usp;
  __shared__ float rinv[16];

  const int qt = blockIdx.x;
  const int hh = blockIdx.y;
  const int bb = blockIdx.z;
  const int tid  = threadIdx.x;
  const int lane = tid & 31;
  const int wave = tid >> 5;
  const int lo = lane & 15;
  const int hi = lane >> 4;
  const float scale = 0.08838834764831845f;      // 1/sqrt(128)
  const float slope = exp2f(-(float)(hh + 1));   // 0.5^(h+1)

  // Phase 0: stage Q tile as bf16 pairs
  {
    int r  = tid >> 4;
    int d0 = (tid & 15) * 8;
    int q  = qt * 16 + r; if (q > T_ - 1) q = T_ - 1;
    const float* src = qkv + ((size_t)(bb * TA_ + 1 + q) * (3 * D_)) + hh * HD_ + d0;
    float4 v0 = ((const float4*)src)[0];
    float4 v1 = ((const float4*)src)[1];
    int pbase = r * 68 + (d0 >> 1);
    uqv.qs[pbase + 0] = pk2(v0.x, v0.y);
    uqv.qs[pbase + 1] = pk2(v0.z, v0.w);
    uqv.qs[pbase + 2] = pk2(v1.x, v1.y);
    uqv.qs[pbase + 3] = pk2(v1.z, v1.w);
  }
  __syncthreads();

  // Phase 1: S = Q*K^T * scale + alibi (waves round-robin over kv tiles)
  for (int jt = wave; jt < NKT; jt += 8) {
    int jcol = jt * 16 + lo; if (jcol > TA_ - 1) jcol = TA_ - 1;  // clamp
    const float* kbase =
        qkv + ((size_t)(bb * TA_ + jcol) * (3 * D_)) + D_ + hh * HD_;
    v8f acc = (v8f)0.f;
#pragma unroll
    for (int kk = 0; kk < 4; kk++) {
      U16 au, bu;
#pragma unroll
      for (int v = 0; v < 4; v++) {
        au.u[v]     = uqv.qs[lo * 68 + kk * 16 + hi * 8 + v];
        au.u[4 + v] = uqv.qs[lo * 68 + kk * 16 + hi * 8 + 4 + v];
      }
      const float4* kp4 = (const float4*)(kbase + kk * 32 + hi * 16);
      float4 f0 = kp4[0], f1 = kp4[1], f2 = kp4[2], f3 = kp4[3];
      bu.u[0] = pk2(f0.x, f0.y); bu.u[1] = pk2(f0.z, f0.w);
      bu.u[2] = pk2(f1.x, f1.y); bu.u[3] = pk2(f1.z, f1.w);
      bu.u[4] = pk2(f2.x, f2.y); bu.u[5] = pk2(f2.z, f2.w);
      bu.u[6] = pk2(f3.x, f3.y); bu.u[7] = pk2(f3.z, f3.w);
      acc = __builtin_amdgcn_wmma_f32_16x16x32_bf16(
          false, au.v, false, bu.v, (short)0, acc, false, false);
    }
#pragma unroll
    for (int r = 0; r < 8; r++) {
      int m  = r + hi * 8;
      int q  = qt * 16 + m;
      int jj = jt * 16 + lo;
      float val;
      if (jj >= TA_) {
        val = -1e30f;                       // pad columns -> zero prob
      } else if (jj == 0) {
        val = acc[r] * scale;               // adapter column, bias 0
      } else {
        int di = q - (jj - 1);
        int steps = (di >= 0) ? (di / PERIOD_) : ((-di - 1) / PERIOD_);
        val = acc[r] * scale - slope * (float)steps;
      }
      usp.s[m * SCOLS + jj] = val;
    }
  }
  __syncthreads();

  // Phase 2: softmax; write bf16 P pairs in place over S (wave-private rows,
  // all reads precede all writes in wave lockstep -> safe)
  for (int row = wave * 2; row < wave * 2 + 2; row++) {
    float v0[10], v1[10];
    float mx = -1e30f;
#pragma unroll
    for (int i = 0; i < 10; i++) {
      int c0 = lane * 2 + i * 64;
      bool ok = c0 < SCOLS;
      int cc = ok ? c0 : 0;
      float a  = usp.s[row * SCOLS + cc];
      float b2 = usp.s[row * SCOLS + cc + 1];
      v0[i] = ok ? a : -1e30f;
      v1[i] = ok ? b2 : -1e30f;
      mx = fmaxf(mx, fmaxf(v0[i], v1[i]));
    }
#pragma unroll
    for (int o = 16; o >= 1; o >>= 1) mx = fmaxf(mx, __shfl_xor(mx, o, 32));
    float sum = 0.f;
#pragma unroll
    for (int i = 0; i < 10; i++) {
      int c0 = lane * 2 + i * 64;
      if (c0 < SCOLS) {
        float e0 = __expf(v0[i] - mx);
        float e1 = __expf(v1[i] - mx);
        sum += e0 + e1;
        usp.p[row * SCOLS + (c0 >> 1)] = pk2(e0, e1);
      }
    }
#pragma unroll
    for (int o = 16; o >= 1; o >>= 1) sum += __shfl_xor(sum, o, 32);
    if (lane == 0) rinv[row] = 1.f / sum;
  }

  // Phase 3: O = P @ V with V staged to LDS transposed+pair-packed (dbl buf)
  auto stageV = [&](int kt, int bufi) {
    int jp = tid >> 4;                 // 0..15 -> j pair
    int d0 = (tid & 15) * 8;           // 0..120
    int j0 = kt * 32 + jp * 2;
    int j0c = j0     < TA_ ? j0     : TA_ - 1;
    int j1c = j0 + 1 < TA_ ? j0 + 1 : TA_ - 1;
    const float* r0 = qkv + ((size_t)(bb * TA_ + j0c)) * (3 * D_) + 2 * D_ + hh * HD_ + d0;
    const float* r1 = qkv + ((size_t)(bb * TA_ + j1c)) * (3 * D_) + 2 * D_ + hh * HD_ + d0;
    float4 x0 = ((const float4*)r0)[0], x1 = ((const float4*)r0)[1];
    float4 y0 = ((const float4*)r1)[0], y1 = ((const float4*)r1)[1];
    uint32_t* vt = uqv.vt[bufi];
    vt[(d0 + 0) * 17 + jp] = pk2(x0.x, y0.x);
    vt[(d0 + 1) * 17 + jp] = pk2(x0.y, y0.y);
    vt[(d0 + 2) * 17 + jp] = pk2(x0.z, y0.z);
    vt[(d0 + 3) * 17 + jp] = pk2(x0.w, y0.w);
    vt[(d0 + 4) * 17 + jp] = pk2(x1.x, y1.x);
    vt[(d0 + 5) * 17 + jp] = pk2(x1.y, y1.y);
    vt[(d0 + 6) * 17 + jp] = pk2(x1.z, y1.z);
    vt[(d0 + 7) * 17 + jp] = pk2(x1.w, y1.w);
  };

  stageV(0, 0);
  {
    int n0 = wave * 16;   // this wave's 16 output dims of the head
    v8f acc = (v8f)0.f;
    int buf = 0;
    for (int kt = 0; kt < 19; kt++) {
      __syncthreads();
      if (kt + 1 < 19) stageV(kt + 1, buf ^ 1);
      U16 au, bu;
#pragma unroll
      for (int v = 0; v < 4; v++) {
        au.u[v]     = usp.p[lo * SCOLS + kt * 16 + hi * 4 + v];
        au.u[4 + v] = usp.p[lo * SCOLS + kt * 16 + 8 + hi * 4 + v];
      }
      const uint32_t* vt = uqv.vt[buf];
#pragma unroll
      for (int v = 0; v < 8; v++) bu.u[v] = vt[(n0 + lo) * 17 + (v >> 2) * 8 + hi * 4 + (v & 3)];
      acc = __builtin_amdgcn_wmma_f32_16x16x32_bf16(
          false, au.v, false, bu.v, (short)0, acc, false, false);
      buf ^= 1;
    }
#pragma unroll
    for (int r = 0; r < 8; r++) {
      int m = r + hi * 8;
      int q = qt * 16 + m;
      if (q < T_)
        out[((size_t)(bb * T_ + q)) * D_ + hh * HD_ + n0 + lo] = acc[r] * rinv[m];
    }
  }
}

// ---------------------------------------------------------------------------
// Cross-attention: eye(T,S) mask + adapter column -> exactly 2 keys/query.
// ---------------------------------------------------------------------------
__global__ __launch_bounds__(256)
void cross_attn(const float* __restrict__ qb, const float* __restrict__ kvb,
                float* __restrict__ out) {
  const int q  = blockIdx.x;
  const int b  = blockIdx.y;
  const int h  = threadIdx.x >> 5;
  const int ln = threadIdx.x & 31;
  const float scale = 0.08838834764831845f;
  const float* qp  = qb  + ((size_t)(b * T_ + q)) * D_ + h * HD_;
  const float* k0p = kvb + ((size_t)(b * TA_ + 0)) * (2 * D_) + h * HD_;
  const float* kqp = kvb + ((size_t)(b * TA_ + 1 + q)) * (2 * D_) + h * HD_;
  float4 q4  = *(const float4*)(qp  + ln * 4);
  float4 k04 = *(const float4*)(k0p + ln * 4);
  float4 kq4 = *(const float4*)(kqp + ln * 4);
  float s0 = q4.x * k04.x + q4.y * k04.y + q4.z * k04.z + q4.w * k04.w;
  float s1 = q4.x * kq4.x + q4.y * kq4.y + q4.z * kq4.z + q4.w * kq4.w;
#pragma unroll
  for (int o = 16; o >= 1; o >>= 1) {
    s0 += __shfl_xor(s0, o, 32);
    s1 += __shfl_xor(s1, o, 32);
  }
  s0 *= scale; s1 *= scale;
  float mx = fmaxf(s0, s1);
  float e0 = __expf(s0 - mx), e1 = __expf(s1 - mx);
  float inv = 1.f / (e0 + e1);
  float w0 = e0 * inv, w1 = e1 * inv;
  float4 v04 = *(const float4*)(k0p + D_ + ln * 4);
  float4 vq4 = *(const float4*)(kqp + D_ + ln * 4);
  float4 o4;
  o4.x = w0 * v04.x + w1 * vq4.x;
  o4.y = w0 * v04.y + w1 * vq4.y;
  o4.z = w0 * v04.z + w1 * vq4.z;
  o4.w = w0 * v04.w + w1 * vq4.w;
  *(float4*)(out + ((size_t)(b * T_ + q)) * D_ + h * HD_ + ln * 4) = o4;
}

// ---------------------------------------------------------------------------
// Fused residual + LayerNorm: h = LN(h + delta) * g + b  (row = 1024)
// ---------------------------------------------------------------------------
__global__ __launch_bounds__(256)
void add_ln(float* __restrict__ hbuf, const float* __restrict__ delta,
            const float* __restrict__ g, const float* __restrict__ bta) {
  __shared__ float red[256];
  const int row = blockIdx.x;
  const int tid = threadIdx.x;
  float xv[4];
  float s = 0.f;
#pragma unroll
  for (int i = 0; i < 4; i++) {
    int c = tid + i * 256;
    xv[i] = hbuf[(size_t)row * D_ + c] + delta[(size_t)row * D_ + c];
    s += xv[i];
  }
  red[tid] = s; __syncthreads();
  for (int o = 128; o > 0; o >>= 1) { if (tid < o) red[tid] += red[tid + o]; __syncthreads(); }
  float mu = red[0] * (1.f / (float)D_);
  __syncthreads();
  float s2 = 0.f;
#pragma unroll
  for (int i = 0; i < 4; i++) { float d = xv[i] - mu; s2 += d * d; }
  red[tid] = s2; __syncthreads();
  for (int o = 128; o > 0; o >>= 1) { if (tid < o) red[tid] += red[tid + o]; __syncthreads(); }
  float rs = rsqrtf(red[0] * (1.f / (float)D_) + 1e-5f);
#pragma unroll
  for (int i = 0; i < 4; i++) {
    int c = tid + i * 256;
    hbuf[(size_t)row * D_ + c] = (xv[i] - mu) * rs * g[c] + bta[c];
  }
}

// ---------------------------------------------------------------------------
// Small/naive kernels
// ---------------------------------------------------------------------------
__global__ void dense_naive(const float* __restrict__ A, const float* __restrict__ W,
                            const float* __restrict__ bias, float* __restrict__ C,
                            int M, int N, int K, int act) {
  int idx = blockIdx.x * blockDim.x + threadIdx.x;
  if (idx >= M * N) return;
  int m = idx / N, n = idx % N;
  const float* a = A + (size_t)m * K;
  const float* w = W + (size_t)n * K;
  float s = bias ? bias[n] : 0.f;
  for (int k = 0; k < K; k++) s += a[k] * w[k];
  if (act == 1) s = s > 0.f ? s : 0.f;                 // relu
  if (act == 2) s = s / (1.f + __expf(-s));            // silu
  C[idx] = s;
}

__global__ void time_sinusoid(const int* __restrict__ ts, float* __restrict__ o) {
  int idx = blockIdx.x * blockDim.x + threadIdx.x;
  if (idx >= B_ * D_) return;
  int b = idx / D_, i = idx % D_;
  const int half = D_ / 2;
  int j = (i < half) ? i : i - half;
  float ex = __expf(-logf(10000.f) * (float)j / (float)(half - 1));
  float e = (float)ts[b] * ex;
  o[idx] = (i < half) ? sinf(e) : cosf(e);
}

__global__ void in_proj(const float* __restrict__ x, const float* __restrict__ W,
                        const float* __restrict__ bias, const float* __restrict__ pe,
                        float* __restrict__ hout) {
  int idx = blockIdx.x * blockDim.x + threadIdx.x;
  if (idx >= B_ * T_ * D_) return;
  int n = idx % D_;
  int m = idx / D_;
  int t = m % T_;
  const float* a = x + (size_t)m * IN_;
  const float* w = W + (size_t)n * IN_;
  float s = bias[n] + pe[(size_t)t * D_ + n];
#pragma unroll
  for (int k = 0; k < IN_; k++) s += a[k] * w[k];
  hout[idx] = s;
}

__global__ void concat_adapter(const float* __restrict__ adapter,
                               const float* __restrict__ rows,
                               float* __restrict__ out, int nrows) {
  size_t idx = (size_t)blockIdx.x * blockDim.x + threadIdx.x;
  size_t total = (size_t)B_ * (nrows + 1) * D_;
  if (idx >= total) return;
  int d = (int)(idx % D_);
  size_t rr = idx / D_;
  int r = (int)(rr % (nrows + 1));
  int b = (int)(rr / (nrows + 1));
  out[idx] = (r == 0) ? adapter[(size_t)b * D_ + d]
                      : rows[((size_t)b * nrows + (r - 1)) * D_ + d];
}

// ---------------------------------------------------------------------------
// Host orchestration
// ---------------------------------------------------------------------------
extern "C" void kernel_launch(void* const* d_in, const int* in_sizes, int n_in,
                              void* d_out, int out_size, void* d_ws, size_t ws_size,
                              hipStream_t stream) {
  const float* x       = (const float*)d_in[0];
  const float* memory  = (const float*)d_in[1];
  const int*   tsteps  = (const int*)  d_in[2];
  const float* pe      = (const float*)d_in[3];
  // d_in[4] = alibi (recomputed analytically in-kernel)
  const float* W_in    = (const float*)d_in[5];
  const float* b_in    = (const float*)d_in[6];
  const float* te_W1   = (const float*)d_in[7];
  const float* te_b1   = (const float*)d_in[8];
  const float* te_W2   = (const float*)d_in[9];
  const float* te_b2   = (const float*)d_in[10];
  const float* sa_Wqkv = (const float*)d_in[11];
  const float* sa_bqkv = (const float*)d_in[12];
  const float* sa_Wo   = (const float*)d_in[13];
  const float* sa_bo   = (const float*)d_in[14];
  const float* ca_Wqkv = (const float*)d_in[15];
  const float* ca_bqkv = (const float*)d_in[16];
  const float* ca_Wo   = (const float*)d_in[17];
  const float* ca_bo   = (const float*)d_in[18];
  const float* ff_W1   = (const float*)d_in[19];
  const float* ff_b1   = (const float*)d_in[20];
  const float* ff_W2   = (const float*)d_in[21];
  const float* ff_b2   = (const float*)d_in[22];
  const float* ln1_g   = (const float*)d_in[23];
  const float* ln2_g   = (const float*)d_in[24];
  const float* ln3_g   = (const float*)d_in[25];
  const float* ln1_b   = (const float*)d_in[26];
  const float* ln2_b   = (const float*)d_in[27];
  const float* ln3_b   = (const float*)d_in[28];
  const float* W_out   = (const float*)d_in[29];
  const float* b_out   = (const float*)d_in[30];
  (void)in_sizes; (void)n_in; (void)out_size; (void)ws_size;

  size_t off = 0;
  auto alloc = [&](size_t nf) {
    float* p = (float*)d_ws + off;
    off += (nf + 63) & ~(size_t)63;
    return p;
  };
  float* temb0   = alloc((size_t)B_ * D_);
  float* temb1   = alloc((size_t)B_ * D_);
  float* adapter = alloc((size_t)B_ * D_);
  float* h       = alloc((size_t)B_ * T_ * D_);
  float* xa      = alloc((size_t)B_ * TA_ * D_);
  float* ma      = alloc((size_t)B_ * TA_ * D_);
  float* qkv     = alloc((size_t)B_ * TA_ * 3 * D_);   // reused for cross q/kv
  float* attn    = alloc((size_t)B_ * T_ * D_);
  float* delta   = alloc((size_t)B_ * T_ * D_);
  float* ffb     = alloc((size_t)B_ * T_ * DFF_);

  const int MT  = B_ * T_;    // 9600
  const int MTA = B_ * TA_;   // 9616

  // Time-embedding MLP -> adapter token
  time_sinusoid<<<(B_ * D_ + 255) / 256, 256, 0, stream>>>(tsteps, temb0);
  dense_naive<<<(B_ * D_ + 255) / 256, 256, 0, stream>>>(temb0, te_W1, te_b1, temb1, B_, D_, D_, 2);
  dense_naive<<<(B_ * D_ + 255) / 256, 256, 0, stream>>>(temb1, te_W2, te_b2, adapter, B_, D_, D_, 0);

  // Input projection + positional encoding
  in_proj<<<(MT * D_ + 255) / 256, 256, 0, stream>>>(x, W_in, b_in, pe, h);

  // memory + adapter (layer-invariant)
  concat_adapter<<<(int)(((size_t)MTA * D_ + 255) / 256), 256, 0, stream>>>(adapter, memory, ma, S_);

  for (int l = 0; l < L_; l++) {
    // ---- self-attention ----
    concat_adapter<<<(int)(((size_t)MTA * D_ + 255) / 256), 256, 0, stream>>>(adapter, h, xa, T_);
    gemm_bf16<<<dim3(3 * D_ / BN, (MTA + BM - 1) / BM), 256, 0, stream>>>(
        xa, sa_Wqkv + (size_t)l * 3 * D_ * D_, sa_bqkv + (size_t)l * 3 * D_,
        qkv, MTA, 3 * D_, D_, 0);
    self_attn<<<dim3(NKT, H_, B_), 256, 0, stream>>>(qkv, attn);
    gemm_bf16<<<dim3(D_ / BN, (MT + BM - 1) / BM), 256, 0, stream>>>(
        attn, sa_Wo + (size_t)l * D_ * D_, sa_bo + (size_t)l * D_,
        delta, MT, D_, D_, 0);
    add_ln<<<MT, 256, 0, stream>>>(h, delta, ln1_g + (size_t)l * D_, ln1_b + (size_t)l * D_);

    // ---- cross-attention (exact 2-key form) ----
    const float* Wq_ca  = ca_Wqkv + (size_t)l * 3 * D_ * D_;
    const float* Wkv_ca = Wq_ca + (size_t)D_ * D_;
    float* qb  = qkv;                          // [MT, D]
    float* kvb = qkv + (size_t)MTA * D_;       // [MTA, 2D]
    gemm_bf16<<<dim3(D_ / BN, (MT + BM - 1) / BM), 256, 0, stream>>>(
        h, Wq_ca, ca_bqkv + (size_t)l * 3 * D_, qb, MT, D_, D_, 0);
    gemm_bf16<<<dim3(2 * D_ / BN, (MTA + BM - 1) / BM), 256, 0, stream>>>(
        ma, Wkv_ca, ca_bqkv + (size_t)l * 3 * D_ + D_, kvb, MTA, 2 * D_, D_, 0);
    cross_attn<<<dim3(T_, B_), 256, 0, stream>>>(qb, kvb, attn);
    gemm_bf16<<<dim3(D_ / BN, (MT + BM - 1) / BM), 256, 0, stream>>>(
        attn, ca_Wo + (size_t)l * D_ * D_, ca_bo + (size_t)l * D_,
        delta, MT, D_, D_, 0);
    add_ln<<<MT, 256, 0, stream>>>(h, delta, ln2_g + (size_t)l * D_, ln2_b + (size_t)l * D_);

    // ---- FFN ----
    gemm_bf16<<<dim3(DFF_ / BN, (MT + BM - 1) / BM), 256, 0, stream>>>(
        h, ff_W1 + (size_t)l * DFF_ * D_, ff_b1 + (size_t)l * DFF_,
        ffb, MT, DFF_, D_, 1);
    gemm_bf16<<<dim3(D_ / BN, (MT + BM - 1) / BM), 256, 0, stream>>>(
        ffb, ff_W2 + (size_t)l * D_ * DFF_, ff_b2 + (size_t)l * D_,
        delta, MT, D_, DFF_, 0);
    add_ln<<<MT, 256, 0, stream>>>(h, delta, ln3_g + (size_t)l * D_, ln3_b + (size_t)l * D_);
  }

  // Output projection (N=52 ragged: naive VALU, ~0.5 GFLOP)
  dense_naive<<<(MT * IN_ + 255) / 256, 256, 0, stream>>>(
      h, W_out, b_out, (float*)d_out, MT, IN_, D_, 0);
}
